// ObliviousDecisionTreeLayer_28346784154074
// MI455X (gfx1250) — matrix operand order repaired and checked by reading, not compiled, not run on hardware
//
#include <hip/hip_runtime.h>

#define BB   4096
#define FF   2048
#define TT   64
#define DD   6
#define LL   64
#define TD   (TT*DD)      // 384
#define KC   64
#define NKC  (FF/KC)      // 32
#define BLK_M 64
#define BLK_N 96

typedef __attribute__((ext_vector_type(16))) __bf16 v16bf;
typedef __attribute__((ext_vector_type(8)))  float  v8f;
typedef int vi4 __attribute__((vector_size(16)));   // matches builtin's V4i

struct alignas(16) U4 { unsigned a, b, c, d; };
struct alignas(16) F4 { float x, y, z, w; };

union BF16x16 { v16bf v; U4 q[2]; };

__device__ __forceinline__ unsigned short bf16u(float f) {
  unsigned u = __builtin_bit_cast(unsigned, f);
  u = u + 0x7FFFu + ((u >> 16) & 1u);     // round-to-nearest-even
  return (unsigned short)(u >> 16);
}
__device__ __forceinline__ unsigned pk2(float lo, float hi) {
  return (unsigned)bf16u(lo) | ((unsigned)bf16u(hi) << 16);
}

// CDNA5 async copy: global -> LDS, 16B per lane, tracked by ASYNCcnt.
__device__ __forceinline__ void async_b128(const void* g, void* l) {
  __builtin_amdgcn_global_load_async_to_lds_b128(
      (__attribute__((address_space(1))) vi4*)g,
      (__attribute__((address_space(3))) vi4*)l, 0, 0);
}

// ---------------------------------------------------------------------------
// Kernel 0: stream-convert x (f32) -> bf16 in workspace (memory-bound).
// ---------------------------------------------------------------------------
__global__ __launch_bounds__(256)
void k_cvt_x(const float* __restrict__ x, unsigned short* __restrict__ xb) {
  const size_t i = ((size_t)blockIdx.x * 256 + threadIdx.x) * 8;
  const F4* p = (const F4*)(x + i);
  F4 a = p[0], b = p[1];
  U4 u;
  u.a = pk2(a.x, a.y); u.b = pk2(a.z, a.w);
  u.c = pk2(b.x, b.y); u.d = pk2(b.z, b.w);
  *(U4*)(xb + i) = u;
}

// ---------------------------------------------------------------------------
// Kernel 1: row softmax over F, f32 -> bf16, layout fwb[n][k], n = t*6+d
// ---------------------------------------------------------------------------
__global__ __launch_bounds__(256)
void k_softmax(const float* __restrict__ fs, unsigned short* __restrict__ fwb) {
  __shared__ float red[256];
  const int n   = blockIdx.x;            // 0..383
  const int tid = threadIdx.x;
  const float* row = fs + (size_t)n * FF;

  float v[8];
  float mx = -3.4e38f;
#pragma unroll
  for (int i = 0; i < 8; ++i) { v[i] = row[tid + i * 256]; mx = fmaxf(mx, v[i]); }
  red[tid] = mx; __syncthreads();
  for (int s = 128; s > 0; s >>= 1) {
    if (tid < s) red[tid] = fmaxf(red[tid], red[tid + s]);
    __syncthreads();
  }
  mx = red[0]; __syncthreads();

  float sum = 0.f;
#pragma unroll
  for (int i = 0; i < 8; ++i) { v[i] = __expf(v[i] - mx); sum += v[i]; }
  red[tid] = sum; __syncthreads();
  for (int s = 128; s > 0; s >>= 1) {
    if (tid < s) red[tid] += red[tid + s];
    __syncthreads();
  }
  const float inv = 1.0f / red[0];
#pragma unroll
  for (int i = 0; i < 8; ++i)
    fwb[(size_t)n * FF + tid + i * 256] = bf16u(v[i] * inv);
}

// ---------------------------------------------------------------------------
// Kernel 2: bf16-WMMA GEMM (selected = x @ fw^T) via async-LDS double buffer,
//           fused sigmoid + oblivious-tree combine epilogue.
// K-chunk = 64: every thread issues exactly 5 asyncs/chunk (uniform waits),
// each barrier round covers 6 WMMAs, and the two K-substeps accumulate into
// independent chains (no WMMA->WMMA register shuffles / hazard nops).
// ---------------------------------------------------------------------------
__global__ __launch_bounds__(256)
void k_gemm_tree(const unsigned short* __restrict__ xb,
                 const unsigned short* __restrict__ fwb,
                 const float* __restrict__ thr,
                 const float* __restrict__ leaf,
                 float* __restrict__ out) {
  // LDS: x tiles [2][64 rows * 9 U4 (144B pitch)], fw tiles [2][96 rows * 9 U4];
  // epilogue reuses the same memory as a 64x97-float accumulator board.
  __shared__ U4 smem[2880];               // 46080 B
  U4*    ldsx = smem;                     // + buf*576
  U4*    ldsw = smem + 1152;              // + buf*864
  float* lacc = (float*)smem;             // 64*97 floats

  const int tid  = threadIdx.x;
  const int mb   = blockIdx.x * BLK_M;    // batch-row base
  const int nb   = blockIdx.y * BLK_N;    // (t*6+d) column base
  const int lane = tid & 31;
  const int wave = tid >> 5;              // 0..7
  const int wm   = wave & 3;              // M wave: rows wm*16..
  const int wn   = wave >> 2;             // N wave: cols wn*48..
  const int lr   = lane & 15;
  const int lh   = lane >> 4;

  // 16B pieces per chunk: x = 64 rows x 8 = 512 (2/thread);
  // fw = 96 rows x 8 = 768 (3/thread). Uniform 5 asyncs per thread.
  auto issue = [&](int kc) {
    const int buf = kc & 1;
#pragma unroll
    for (int i = 0; i < 2; ++i) {
      const int p = tid + i * 256, row = p >> 3, pc = p & 7;
      async_b128(xb + (size_t)(mb + row) * FF + kc * KC + pc * 8,
                 (char*)(ldsx + buf * 576 + row * 9) + pc * 16);
    }
#pragma unroll
    for (int i = 0; i < 3; ++i) {
      const int p = tid + i * 256, row = p >> 3, pc = p & 7;
      async_b128(fwb + (size_t)(nb + row) * FF + kc * KC + pc * 8,
                 (char*)(ldsw + buf * 864 + row * 9) + pc * 16);
    }
  };

  v8f acc[2][3] = {};                     // independent chain per K-substep

  issue(0);
  for (int kc = 0; kc < NKC; ++kc) {
    if (kc + 1 < NKC) {
      __syncthreads();                    // all reads of buf (kc+1)&1 done
      issue(kc + 1);
      __builtin_amdgcn_s_wait_asynccnt(5);  // drain chunk kc, keep kc+1 in flight
    } else {
      __builtin_amdgcn_s_wait_asynccnt(0);
    }
    __syncthreads();                      // publish chunk kc tile

    const int buf = kc & 1;
    const U4* bx = ldsx + buf * 576 + (wm * 16 + lr) * 9;
    const U4* bw = ldsw + buf * 864 + (wn * 48 + lr) * 9 + lh * 2;
#pragma unroll
    for (int ks = 0; ks < 2; ++ks) {      // two K=32 sub-steps
      // A fragment (16x32 bf16): lane half lh holds 32 contiguous bytes at lh*32
      // (K = lh*16..lh*16+15) of its row.
      BF16x16 fa, fb0, fb1, fb2;
      fa.q[0] = bx[ks * 4 + lh * 2];
      fa.q[1] = bx[ks * 4 + lh * 2 + 1];
      // B fragment (32x16 bf16): lane half lh holds 32 contiguous bytes at lh*32.
      fb0.q[0] = bw[ks * 4];       fb0.q[1] = bw[ks * 4 + 1];
      fb1.q[0] = bw[ks * 4 + 144]; fb1.q[1] = bw[ks * 4 + 145];  // +16 rows * 9 U4
      fb2.q[0] = bw[ks * 4 + 288]; fb2.q[1] = bw[ks * 4 + 289];

      acc[ks][0] = __builtin_amdgcn_wmma_f32_16x16x32_bf16(
          false, fa.v, false, fb0.v, (short)0, acc[ks][0], false, false);
      acc[ks][1] = __builtin_amdgcn_wmma_f32_16x16x32_bf16(
          false, fa.v, false, fb1.v, (short)0, acc[ks][1], false, false);
      acc[ks][2] = __builtin_amdgcn_wmma_f32_16x16x32_bf16(
          false, fa.v, false, fb2.v, (short)0, acc[ks][2], false, false);
    }
  }
  __syncthreads();

  // Reduce the two chains, then stage to LDS:
  // D element (j, lane) is M = j + 8*lh, N = lr.
  {
    const v8f acc0 = acc[0][0] + acc[1][0];
    const v8f acc1 = acc[0][1] + acc[1][1];
    const v8f acc2 = acc[0][2] + acc[1][2];
    const int mrow = wm * 16 + 8 * lh;
    const int ncol = wn * 48 + lr;
#pragma unroll
    for (int j = 0; j < 8; ++j) {
      lacc[(mrow + j) * 97 + ncol]      = acc0[j];
      lacc[(mrow + j) * 97 + ncol + 16] = acc1[j];
      lacc[(mrow + j) * 97 + ncol + 32] = acc2[j];
    }
  }
  __syncthreads();

  // Tree combine: 64 rows x 16 trees per WG -> 1024 tasks, 4 per thread.
  // Butterfly contraction over the 6 decision bits: 2*(32+16+8+4+2+1) FMAs.
  const int t0 = blockIdx.y * 16;
#pragma unroll
  for (int i = 0; i < 4; ++i) {
    const int task = tid + i * 256;
    const int m  = task & 63;
    const int tl = task >> 6;            // 0..15
    const int t  = t0 + tl;

    float dec[6];
#pragma unroll
    for (int d = 0; d < 6; ++d) {
      const float s  = lacc[m * 97 + tl * 6 + d];
      const float th = thr[t * 6 + d];
      dec[d] = 1.0f / (1.0f + __expf(th - s));   // sigmoid(s - th)
    }
    const float* lv = leaf + (size_t)t * LL;
    float v[LL];
#pragma unroll
    for (int l = 0; l < LL; ++l) v[l] = lv[l];
#pragma unroll
    for (int lev = 0; lev < 6; ++lev) {          // lev 0 contracts bit0 (d=5)
      const float dd = dec[5 - lev];
      const int n = 32 >> lev;
#pragma unroll
      for (int j = 0; j < n; ++j)
        v[j] = v[2 * j] + dd * (v[2 * j + 1] - v[2 * j]);
    }
    out[(size_t)(mb + m) * TT + t] = v[0];
  }
}

// ---------------------------------------------------------------------------
extern "C" void kernel_launch(void* const* d_in, const int* in_sizes, int n_in,
                              void* d_out, int out_size, void* d_ws, size_t ws_size,
                              hipStream_t stream) {
  const float* x   = (const float*)d_in[0];   // (B, F)
  const float* fs  = (const float*)d_in[1];   // (T, D, F)
  const float* th  = (const float*)d_in[2];   // (T, D)
  const float* lv  = (const float*)d_in[3];   // (T, L)
  float* out = (float*)d_out;                 // (B, T)

  unsigned short* fwb = (unsigned short*)d_ws;        // 384*2048 bf16 = 1.5 MB
  unsigned short* xb  = fwb + (size_t)TD * FF;        // 4096*2048 bf16 = 16 MB

  k_cvt_x<<<dim3((BB * FF) / (256 * 8)), dim3(256), 0, stream>>>(x, xb);
  k_softmax<<<dim3(TD), dim3(256), 0, stream>>>(fs, fwb);
  k_gemm_tree<<<dim3(BB / BLK_M, TD / BLK_N), dim3(256), 0, stream>>>(xb, fwb, th, lv, out);
}
